// EncoderLayer_35897336660579
// MI455X (gfx1250) — matrix operand order repaired and checked by reading, not compiled, not run on hardware
//
#include <hip/hip_runtime.h>

// ---------------------------------------------------------------------------
// Problem constants (match reference)
// ---------------------------------------------------------------------------
#define BB   4
#define SS   1024
#define DD   1024
#define FF   2048
#define HH   16
#define EE   8
#define DKK  64
#define MM   (BB * SS)        // 4096 tokens

#define PF_DIST 8             // k-tile prefetch distance (8 * 1KB per stream)

typedef __attribute__((ext_vector_type(16))) __bf16 v16bf;
typedef __attribute__((ext_vector_type(8)))  float  v8f;

// fp32 -> bf16 round-to-nearest-even
static __device__ __forceinline__ __bf16 f2bf(float f) {
  unsigned u = __builtin_bit_cast(unsigned, f);
  unsigned r = u + 0x7FFFu + ((u >> 16) & 1u);
  unsigned short h = (unsigned short)(r >> 16);
  return __builtin_bit_cast(__bf16, h);
}

static __device__ __forceinline__ v8f wmma_bf16(v16bf a, v16bf b, v8f c) {
  // (neg_a, A, neg_b, B, c_mod, C, reuse_a, reuse_b)
  return __builtin_amdgcn_wmma_f32_16x16x32_bf16(false, a, false, b, (short)0, c,
                                                 false, false);
}

// speculative prefetch (lowers to global_prefetch_b8; dropped if addr invalid)
static __device__ __forceinline__ void pf(const void* p) {
  __builtin_prefetch(p, 0, 1);
}

// ---------------------------------------------------------------------------
// Pack fp32 matrix into CDNA5 WMMA A-fragment layout (16x32 bf16 tiles).
// element [m][k] = src[base + m*rs + k*cs]; base derived from blockIdx.z.
// A layout (ISA 7.12.2): lane<16: j0..7 -> K=0..7, j8..15 -> K=16..23
//                        lane>=16: j0..7 -> K=8..15, j8..15 -> K=24..31
// ---------------------------------------------------------------------------
__global__ void pack_a_kernel(const float* __restrict__ src, __bf16* __restrict__ dst,
                              int Mt, int Kt, int rs, int cs,
                              long zb, long zh, int Hdim) {
  int total = Mt * Kt * 32;
  int t = blockIdx.x * blockDim.x + threadIdx.x;
  if (t >= total) return;
  int z = blockIdx.z;
  long base = (long)(z / Hdim) * zb + (long)(z % Hdim) * zh;
  int lane = t & 31, tile = t >> 5;
  int mt = tile / Kt, kt = tile - mt * Kt;
  int half = lane >> 4;
  int m  = mt * 16 + (lane & 15);
  int k0 = kt * 32 + half * 8;
  int k1 = k0 + 16;
  const float* p = src + base + (long)m * rs;
  v16bf v;
#pragma unroll
  for (int j = 0; j < 8; j++) v[j]     = f2bf(p[(long)(k0 + j) * cs]);
#pragma unroll
  for (int j = 0; j < 8; j++) v[8 + j] = f2bf(p[(long)(k1 + j) * cs]);
  *(v16bf*)(dst + (long)z * total * 16 + (long)tile * 512 + lane * 16) = v;
}

// ---------------------------------------------------------------------------
// Pack fp32 matrix into WMMA B-fragment layout (32x16 bf16 tiles).
// element [k][n] = src[base + k*rs + n*cs].
// B layout: lane = column n (mod 16); lane<16: j0..7 -> K=0..7, j8..15 -> K=16..23; lane>=16: K=8..15, 24..31.
// ---------------------------------------------------------------------------
__global__ void pack_b_kernel(const float* __restrict__ src, __bf16* __restrict__ dst,
                              int Kt, int Nt, int rs, int cs,
                              long zb, long zh, int Hdim) {
  int total = Kt * Nt * 32;
  int t = blockIdx.x * blockDim.x + threadIdx.x;
  if (t >= total) return;
  int z = blockIdx.z;
  long base = (long)(z / Hdim) * zb + (long)(z % Hdim) * zh;
  int lane = t & 31, tile = t >> 5;
  int kt = tile / Nt, nt = tile - kt * Nt;
  int half = lane >> 4;
  int n  = nt * 16 + (lane & 15);
  int k0 = kt * 32 + half * 8;
  const float* p = src + base + (long)n * cs;
  v16bf v;
#pragma unroll
  for (int j = 0; j < 16; j++) v[j] = f2bf(p[(long)(k0 + j + (j >= 8 ? 8 : 0)) * rs]);
  *(v16bf*)(dst + (long)z * total * 16 + (long)tile * 512 + lane * 16) = v;
}

// ---------------------------------------------------------------------------
// Fused SwiGLU GEMM:  out = silu(A@W1 + b1) * (A@W2 + b2),   fp32 out [M,N]
// Block 256 (8 waves, 4x2), block tile 128x64, wave tile 32x32.
// ---------------------------------------------------------------------------
__global__ __launch_bounds__(256) void gemm_swiglu_kernel(
    const __bf16* __restrict__ A, const __bf16* __restrict__ B1,
    const __bf16* __restrict__ B2, const float* __restrict__ bias1,
    const float* __restrict__ bias2, float* __restrict__ out,
    int Ktiles, int Ntiles, int N) {
  int lane = threadIdx.x & 31, wid = threadIdx.x >> 5;
  int waveM = wid >> 1, waveN = wid & 1;
  int mt0 = blockIdx.y * 8 + waveM * 2;
  int nt0 = blockIdx.x * 4 + waveN * 2;
  const v8f zero = {0.f, 0.f, 0.f, 0.f, 0.f, 0.f, 0.f, 0.f};
  v8f acc1[2][2], acc2[2][2];
#pragma unroll
  for (int i = 0; i < 2; i++)
#pragma unroll
    for (int j = 0; j < 2; j++) { acc1[i][j] = zero; acc2[i][j] = zero; }

  for (int kt = 0; kt < Ktiles; kt++) {
    v16bf a0 = *(const v16bf*)(A + ((long)(mt0 + 0) * Ktiles + kt) * 512 + lane * 16);
    v16bf a1 = *(const v16bf*)(A + ((long)(mt0 + 1) * Ktiles + kt) * 512 + lane * 16);
    long bo = ((long)kt * Ntiles + nt0) * 512 + lane * 16;
    v16bf b10 = *(const v16bf*)(B1 + bo);
    v16bf b11 = *(const v16bf*)(B1 + bo + 512);
    v16bf b20 = *(const v16bf*)(B2 + bo);
    v16bf b21 = *(const v16bf*)(B2 + bo + 512);
    // stream prefetch: pull next tiles of the weight/activation streams into cache
    long bpf = ((long)(kt + PF_DIST) * Ntiles + nt0) * 512 + lane * 16;
    pf(B1 + bpf);
    pf(B2 + bpf);
    pf(A + ((long)(mt0 + 0) * Ktiles + (kt + PF_DIST)) * 512 + lane * 16);
    acc1[0][0] = wmma_bf16(a0, b10, acc1[0][0]);
    acc1[0][1] = wmma_bf16(a0, b11, acc1[0][1]);
    acc1[1][0] = wmma_bf16(a1, b10, acc1[1][0]);
    acc1[1][1] = wmma_bf16(a1, b11, acc1[1][1]);
    acc2[0][0] = wmma_bf16(a0, b20, acc2[0][0]);
    acc2[0][1] = wmma_bf16(a0, b21, acc2[0][1]);
    acc2[1][0] = wmma_bf16(a1, b20, acc2[1][0]);
    acc2[1][1] = wmma_bf16(a1, b21, acc2[1][1]);
  }
  int col = lane & 15, half = lane >> 4;
#pragma unroll
  for (int wm = 0; wm < 2; wm++)
#pragma unroll
    for (int wn = 0; wn < 2; wn++) {
      int n = (nt0 + wn) * 16 + col;
      float bb1 = bias1[n], bb2 = bias2[n];
#pragma unroll
      for (int r = 0; r < 8; r++) {
        int m = (mt0 + wm) * 16 + half * 8 + r;
        float a = acc1[wm][wn][r] + bb1;
        float v = acc2[wm][wn][r] + bb2;
        out[(long)m * N + n] = (a / (1.f + __expf(-a))) * v;
      }
    }
}

// ---------------------------------------------------------------------------
// GEMM + bias, optional per-row scale / residual / accumulate:
//   out = (acc? out:0) + (res? res:0) + scale(m) * (A@W + bias)
// Block 256 (8 waves, 4x2), block tile 128x128, wave tile 32x64.
// ---------------------------------------------------------------------------
__global__ __launch_bounds__(256) void gemm_bias_res_kernel(
    const __bf16* __restrict__ A, const __bf16* __restrict__ B,
    const float* __restrict__ bias, const float* __restrict__ res,
    const float* __restrict__ scale, int scale_stride, int acc_flag,
    float* __restrict__ out, int Ktiles, int Ntiles, int N) {
  int lane = threadIdx.x & 31, wid = threadIdx.x >> 5;
  int waveM = wid >> 1, waveN = wid & 1;
  int mt0 = blockIdx.y * 8 + waveM * 2;
  int nt0 = blockIdx.x * 8 + waveN * 4;
  const v8f zero = {0.f, 0.f, 0.f, 0.f, 0.f, 0.f, 0.f, 0.f};
  v8f acc[2][4];
#pragma unroll
  for (int i = 0; i < 2; i++)
#pragma unroll
    for (int j = 0; j < 4; j++) acc[i][j] = zero;

  for (int kt = 0; kt < Ktiles; kt++) {
    v16bf a0 = *(const v16bf*)(A + ((long)(mt0 + 0) * Ktiles + kt) * 512 + lane * 16);
    v16bf a1 = *(const v16bf*)(A + ((long)(mt0 + 1) * Ktiles + kt) * 512 + lane * 16);
    long bo = ((long)kt * Ntiles + nt0) * 512 + lane * 16;
    v16bf b0 = *(const v16bf*)(B + bo);
    v16bf b1 = *(const v16bf*)(B + bo + 512);
    v16bf b2 = *(const v16bf*)(B + bo + 1024);
    v16bf b3 = *(const v16bf*)(B + bo + 1536);
    long bpf = ((long)(kt + PF_DIST) * Ntiles + nt0) * 512 + lane * 16;
    pf(B + bpf);
    pf(B + bpf + 1024);
    pf(A + ((long)(mt0 + 0) * Ktiles + (kt + PF_DIST)) * 512 + lane * 16);
    acc[0][0] = wmma_bf16(a0, b0, acc[0][0]);
    acc[0][1] = wmma_bf16(a0, b1, acc[0][1]);
    acc[0][2] = wmma_bf16(a0, b2, acc[0][2]);
    acc[0][3] = wmma_bf16(a0, b3, acc[0][3]);
    acc[1][0] = wmma_bf16(a1, b0, acc[1][0]);
    acc[1][1] = wmma_bf16(a1, b1, acc[1][1]);
    acc[1][2] = wmma_bf16(a1, b2, acc[1][2]);
    acc[1][3] = wmma_bf16(a1, b3, acc[1][3]);
  }
  int col = lane & 15, half = lane >> 4;
#pragma unroll
  for (int wm = 0; wm < 2; wm++)
#pragma unroll
    for (int wn = 0; wn < 4; wn++) {
      int n = (nt0 + wn) * 16 + col;
      float bb = bias[n];
#pragma unroll
      for (int r = 0; r < 8; r++) {
        int m = (mt0 + wm) * 16 + half * 8 + r;
        float v = acc[wm][wn][r] + bb;
        if (scale) v *= scale[(long)m * scale_stride];
        if (res) v += res[(long)m * N + n];
        long o = (long)m * N + n;
        if (acc_flag) v += out[o];
        out[o] = v;
      }
    }
}

// ---------------------------------------------------------------------------
// Flash-style attention. One wave per 16-query tile, 8 waves/block.
// qA: per-(b,h) A-fragments [S/16][DK/32]; kBt: B-fragments of K^T [DK/32][S/16];
// vB: B-fragments of V [S/32][DK/16]. P transposed via padded LDS scratch.
// ---------------------------------------------------------------------------
#define ATT_STRIDE 40  // padded LDS row stride (bf16) to dodge bank conflicts

__global__ __launch_bounds__(256) void attention_kernel(
    const __bf16* __restrict__ qA, const __bf16* __restrict__ kBt,
    const __bf16* __restrict__ vB, const int* __restrict__ mask,
    float* __restrict__ ctx) {
  __shared__ __bf16 lds[8 * 16 * ATT_STRIDE];
  int lane = threadIdx.x & 31, wid = threadIdx.x >> 5;
  int bh = blockIdx.y;
  int b = bh / HH, h = bh % HH;
  int qtile = blockIdx.x * 8 + wid;
  const int perBH = 65536;  // fragment elements per (b,h) for each of qA/kBt/vB
  const __bf16* qb = qA + (long)bh * perBH;
  const __bf16* kb = kBt + (long)bh * perBH;
  const __bf16* vb = vB + (long)bh * perBH;
  const int Stq = SS / 16;  // 64 key/query 16-tiles

  v16bf qa0 = *(const v16bf*)(qb + ((long)qtile * 2 + 0) * 512 + lane * 16);
  v16bf qa1 = *(const v16bf*)(qb + ((long)qtile * 2 + 1) * 512 + lane * 16);

  const v8f zero = {0.f, 0.f, 0.f, 0.f, 0.f, 0.f, 0.f, 0.f};
  v8f acc[4];
#pragma unroll
  for (int i = 0; i < 4; i++) acc[i] = zero;
  float mrow[8], lrow[8];
#pragma unroll
  for (int r = 0; r < 8; r++) { mrow[r] = -3.0e38f; lrow[r] = 0.f; }

  int col = lane & 15, half = lane >> 4;
  __bf16* myl = lds + wid * 16 * ATT_STRIDE;
  const int* mb = mask + (long)b * SS * SS;
  const float invs = 0.125f;  // 1/sqrt(64)

  for (int c = 0; c < SS / 32; c++) {
    // ---- scores: two 16x16 C tiles (keys c*32..+15 and +16..+31) ----
    v8f s0 = zero, s1 = zero;
    int st0 = c * 2, st1 = c * 2 + 1;
    // prefetch next chunk's K^T / V fragment streams
    pf(kb + ((long)0 * Stq + st0 + 2) * 512 + lane * 16);
    pf(kb + ((long)1 * Stq + st0 + 2) * 512 + lane * 16);
    pf(vb + ((long)(c + 1) * 4) * 512 + lane * 16);
    s0 = wmma_bf16(qa0, *(const v16bf*)(kb + ((long)0 * Stq + st0) * 512 + lane * 16), s0);
    s0 = wmma_bf16(qa1, *(const v16bf*)(kb + ((long)1 * Stq + st0) * 512 + lane * 16), s0);
    s1 = wmma_bf16(qa0, *(const v16bf*)(kb + ((long)0 * Stq + st1) * 512 + lane * 16), s1);
    s1 = wmma_bf16(qa1, *(const v16bf*)(kb + ((long)1 * Stq + st1) * 512 + lane * 16), s1);

    // ---- scale + mask + online softmax (per-row shfl reductions) ----
#pragma unroll
    for (int r = 0; r < 8; r++) {
      int q = qtile * 16 + half * 8 + r;
      int k0 = c * 32 + col, k1 = k0 + 16;
      float v0 = (mb[(long)q * SS + k0] == 0) ? -1e9f : s0[r] * invs;
      float v1 = (mb[(long)q * SS + k1] == 0) ? -1e9f : s1[r] * invs;
      float loc = fmaxf(v0, v1);
#pragma unroll
      for (int sh = 8; sh >= 1; sh >>= 1) loc = fmaxf(loc, __shfl_xor(loc, sh, 32));
      float mnew = fmaxf(mrow[r], loc);
      float corr = __expf(mrow[r] - mnew);
      float e0 = __expf(v0 - mnew);
      float e1 = __expf(v1 - mnew);
      float rs = e0 + e1;
#pragma unroll
      for (int sh = 8; sh >= 1; sh >>= 1) rs += __shfl_xor(rs, sh, 32);
      lrow[r] = lrow[r] * corr + rs;
      mrow[r] = mnew;
      acc[0][r] *= corr; acc[1][r] *= corr; acc[2][r] *= corr; acc[3][r] *= corr;
      myl[(half * 8 + r) * ATT_STRIDE + col]      = f2bf(e0);
      myl[(half * 8 + r) * ATT_STRIDE + 16 + col] = f2bf(e1);
    }
    __syncthreads();
    // ---- read P back as A-fragment (16x32) ----
    v16bf pa;
    {
      const __bf16* pr = myl + col * ATT_STRIDE;
      int kk0 = half * 8, kk1 = 16 + half * 8;
#pragma unroll
      for (int j = 0; j < 8; j++) { pa[j] = pr[kk0 + j]; pa[8 + j] = pr[kk1 + j]; }
    }
    // ---- ctx += P @ V ----
#pragma unroll
    for (int nt = 0; nt < 4; nt++) {
      v16bf vf = *(const v16bf*)(vb + ((long)c * 4 + nt) * 512 + lane * 16);
      acc[nt] = wmma_bf16(pa, vf, acc[nt]);
    }
    __syncthreads();
  }
  // ---- normalize + write ctx (fp32 [B,S,D], head-concatenated) ----
#pragma unroll
  for (int nt = 0; nt < 4; nt++)
#pragma unroll
    for (int r = 0; r < 8; r++) {
      int q = qtile * 16 + half * 8 + r;
      int d = h * DKK + nt * 16 + col;
      ctx[((long)b * SS + q) * DD + d] = acc[nt][r] / lrow[r];
    }
}

// ---------------------------------------------------------------------------
// LayerNorm: one wave per token (8 tokens / 256-thread block)
// ---------------------------------------------------------------------------
__global__ __launch_bounds__(256) void ln_kernel(const float* __restrict__ x,
                                                 const float* __restrict__ g,
                                                 const float* __restrict__ b,
                                                 float* __restrict__ y) {
  int tok = blockIdx.x * 8 + (threadIdx.x >> 5);
  int lane = threadIdx.x & 31;
  const float* row = x + (long)tok * DD;
  float s = 0.f, s2 = 0.f;
  for (int i = lane; i < DD; i += 32) { float v = row[i]; s += v; s2 += v * v; }
#pragma unroll
  for (int sh = 16; sh >= 1; sh >>= 1) { s += __shfl_xor(s, sh, 32); s2 += __shfl_xor(s2, sh, 32); }
  float mean = s * (1.f / DD);
  float var = s2 * (1.f / DD) - mean * mean;
  float inv = rsqrtf(var + 1e-5f);
  float* yr = y + (long)tok * DD;
  for (int i = lane; i < DD; i += 32) yr[i] = (row[i] - mean) * inv * g[i] + b[i];
}

// ---------------------------------------------------------------------------
// MoE gate: softmax over E=8 logits, top-2, renormalize -> comb [M, E]
// One wave per token.
// ---------------------------------------------------------------------------
__global__ __launch_bounds__(256) void gate_kernel(const float* __restrict__ x2,
                                                   const float* __restrict__ gw,
                                                   const float* __restrict__ gb,
                                                   float* __restrict__ comb) {
  int tok = blockIdx.x * 8 + (threadIdx.x >> 5);
  int lane = threadIdx.x & 31;
  float a[EE];
#pragma unroll
  for (int e = 0; e < EE; e++) a[e] = 0.f;
  for (int i = lane; i < DD; i += 32) {
    float xv = x2[(long)tok * DD + i];
    const float* gr = gw + (long)i * EE;
#pragma unroll
    for (int e = 0; e < EE; e++) a[e] += xv * gr[e];
  }
#pragma unroll
  for (int e = 0; e < EE; e++)
#pragma unroll
    for (int sh = 16; sh >= 1; sh >>= 1) a[e] += __shfl_xor(a[e], sh, 32);
  if (lane == 0) {
    float mx = -3e38f;
#pragma unroll
    for (int e = 0; e < EE; e++) { a[e] += gb[e]; mx = fmaxf(mx, a[e]); }
    float sum = 0.f;
#pragma unroll
    for (int e = 0; e < EE; e++) { a[e] = __expf(a[e] - mx); sum += a[e]; }
    float inv = 1.f / sum;
    int i1 = 0;
#pragma unroll
    for (int e = 1; e < EE; e++) if (a[e] > a[i1]) i1 = e;
    int i2 = (i1 == 0) ? 1 : 0;
#pragma unroll
    for (int e = 0; e < EE; e++) if (e != i1 && a[e] > a[i2]) i2 = e;
    float p1 = a[i1] * inv, p2 = a[i2] * inv;
    float nrm = 1.f / (p1 + p2 + 1e-6f);
#pragma unroll
    for (int e = 0; e < EE; e++)
      comb[(long)tok * EE + e] = (e == i1) ? p1 * nrm : ((e == i2) ? p2 * nrm : 0.f);
  }
}

// ---------------------------------------------------------------------------
// Host-side launch helpers
// ---------------------------------------------------------------------------
static inline void packA(const float* s, __bf16* d, int Mt, int Kt, int rs, int cs,
                         int Z, int Hd, long zb, long zh, hipStream_t st) {
  int total = Mt * Kt * 32;
  pack_a_kernel<<<dim3((total + 255) / 256, 1, Z), 256, 0, st>>>(s, d, Mt, Kt, rs, cs, zb, zh, Hd);
}
static inline void packB(const float* s, __bf16* d, int Kt, int Nt, int rs, int cs,
                         int Z, int Hd, long zb, long zh, hipStream_t st) {
  int total = Kt * Nt * 32;
  pack_b_kernel<<<dim3((total + 255) / 256, 1, Z), 256, 0, st>>>(s, d, Kt, Nt, rs, cs, zb, zh, Hd);
}

extern "C" void kernel_launch(void* const* d_in, const int* in_sizes, int n_in,
                              void* d_out, int out_size, void* d_ws, size_t ws_size,
                              hipStream_t stream) {
  (void)in_sizes; (void)n_in; (void)out_size; (void)ws_size;
  const float* x    = (const float*)d_in[0];
  const int*   mask = (const int*)d_in[1];
  const float* wq1 = (const float*)d_in[2],  *bq1 = (const float*)d_in[3];
  const float* wq2 = (const float*)d_in[4],  *bq2 = (const float*)d_in[5];
  const float* wk1 = (const float*)d_in[6],  *bk1 = (const float*)d_in[7];
  const float* wk2 = (const float*)d_in[8],  *bk2 = (const float*)d_in[9];
  const float* wv1 = (const float*)d_in[10], *bv1 = (const float*)d_in[11];
  const float* wv2 = (const float*)d_in[12], *bv2 = (const float*)d_in[13];
  const float* wo  = (const float*)d_in[14], *bo  = (const float*)d_in[15];
  const float* ln1g = (const float*)d_in[16], *ln1b = (const float*)d_in[17];
  const float* ln2g = (const float*)d_in[18], *ln2b = (const float*)d_in[19];
  const float* gw  = (const float*)d_in[20], *gb = (const float*)d_in[21];
  const float* ew1 = (const float*)d_in[22], *eb1 = (const float*)d_in[23];
  const float* ew2 = (const float*)d_in[24], *eb2 = (const float*)d_in[25];
  const float* ew3 = (const float*)d_in[26], *eb3 = (const float*)d_in[27];
  float* out = (float*)d_out;

  // workspace carve-up (~128 MB)
  char* W = (char*)d_ws;
  float*  x2    = (float*)(W + (0ul   << 20));  // 16 MB  LN output
  float*  xAttn = (float*)(W + (16ul  << 20));  // 16 MB  post-attention residual
  float*  tmp   = (float*)(W + (32ul  << 20));  // 32 MB  fp32 GEMM outputs / ctx / H
  float*  comb  = (float*)(W + (64ul  << 20));  //  <1 MB gate combine weights [M,E]
  __bf16* x2a   = (__bf16*)(W + (72ul << 20));  //  8 MB  packed A activations
  __bf16* hA    = (__bf16*)(W + (80ul << 20));  // 16 MB  packed A of MoE hidden
  __bf16* wB1   = (__bf16*)(W + (96ul << 20));  //  4 MB  packed B weight slot 1
  __bf16* wB2   = (__bf16*)(W + (100ul << 20)); //  4 MB  packed B weight slot 2
  __bf16* qA    = (__bf16*)(W + (104ul << 20)); //  8 MB  per-head Q A-frags
  __bf16* kBt   = (__bf16*)(W + (112ul << 20)); //  8 MB  per-head K^T B-frags
  __bf16* vB    = (__bf16*)(W + (120ul << 20)); //  8 MB  per-head V B-frags

  const int Dt = DD / 16, Dk32 = DD / 32;   // 64, 32
  const int Ft = FF / 16, Fk32 = FF / 32;   // 128, 64
  const int Mt = MM / 16;                   // 256

  // ---- pre-norm 1 + pack activations ----
  ln_kernel<<<dim3(MM / 8), 256, 0, stream>>>(x, ln1g, ln1b, x2);
  packA(x2, x2a, Mt, Dk32, DD, 1, 1, 1, 0, 0, stream);

  // ---- Q/K/V SwiGLU projections, then pack into attention operand layouts ----
  const float* w1s[3] = {wq1, wk1, wv1};
  const float* b1s[3] = {bq1, bk1, bv1};
  const float* w2s[3] = {wq2, wk2, wv2};
  const float* b2s[3] = {bq2, bk2, bv2};
  for (int p = 0; p < 3; p++) {
    packB(w1s[p], wB1, Dk32, Dt, DD, 1, 1, 1, 0, 0, stream);
    packB(w2s[p], wB2, Dk32, Dt, DD, 1, 1, 1, 0, 0, stream);
    gemm_swiglu_kernel<<<dim3(DD / 64, MM / 128), 256, 0, stream>>>(
        x2a, wB1, wB2, b1s[p], b2s[p], tmp, Dk32, Dt, DD);
    if (p == 0)       // Q as A-fragments per (b,h): [S/16][DK/32]
      packA(tmp, qA, SS / 16, DKK / 32, DD, 1, BB * HH, HH, (long)SS * DD, DKK, stream);
    else if (p == 1)  // K^T as B-fragments per (b,h): rows=d (rs=1), cols=s (cs=D)
      packB(tmp, kBt, DKK / 32, SS / 16, 1, DD, BB * HH, HH, (long)SS * DD, DKK, stream);
    else              // V as B-fragments per (b,h): rows=s (rs=D), cols=d (cs=1)
      packB(tmp, vB, SS / 32, DKK / 16, DD, 1, BB * HH, HH, (long)SS * DD, DKK, stream);
  }

  // ---- attention (ctx -> tmp) ----
  attention_kernel<<<dim3(SS / 128, BB * HH), 256, 0, stream>>>(qA, kBt, vB, mask, tmp);

  // ---- out-projection + residual: xAttn = x + ctx@wo + bo ----
  packA(tmp, x2a, Mt, Dk32, DD, 1, 1, 1, 0, 0, stream);
  packB(wo, wB1, Dk32, Dt, DD, 1, 1, 1, 0, 0, stream);
  gemm_bias_res_kernel<<<dim3(DD / 128, MM / 128), 256, 0, stream>>>(
      x2a, wB1, bo, x, nullptr, 0, 0, xAttn, Dk32, Dt, DD);

  // ---- pre-norm 2 + gate + pack ----
  ln_kernel<<<dim3(MM / 8), 256, 0, stream>>>(xAttn, ln2g, ln2b, x2);
  gate_kernel<<<dim3(MM / 8), 256, 0, stream>>>(x2, gw, gb, comb);
  packA(x2, x2a, Mt, Dk32, DD, 1, 1, 1, 0, 0, stream);

  // ---- MoE: 8 experts, comb-weighted accumulation into d_out ----
  for (int e = 0; e < EE; e++) {
    packB(ew1 + (long)e * DD * FF, wB1, Dk32, Ft, FF, 1, 1, 1, 0, 0, stream);
    packB(ew3 + (long)e * DD * FF, wB2, Dk32, Ft, FF, 1, 1, 1, 0, 0, stream);
    gemm_swiglu_kernel<<<dim3(FF / 64, MM / 128), 256, 0, stream>>>(
        x2a, wB1, wB2, eb1 + (long)e * FF, eb3 + (long)e * FF, tmp, Dk32, Ft, FF);
    packA(tmp, hA, Mt, Fk32, FF, 1, 1, 1, 0, 0, stream);
    packB(ew2 + (long)e * FF * DD, wB1, Fk32, Dt, DD, 1, 1, 1, 0, 0, stream);
    // expert 0 seeds d_out with residual xAttn; others accumulate
    gemm_bias_res_kernel<<<dim3(DD / 128, MM / 128), 256, 0, stream>>>(
        hA, wB1, eb2 + (long)e * DD, (e == 0) ? xAttn : nullptr,
        comb + e, EE, (e == 0) ? 0 : 1, out, Fk32, Dt, DD);
  }
}
